// AttentionBranch_56143812494095
// MI455X (gfx1250) — matrix operand order repaired and checked by reading, not compiled, not run on hardware
//
#include <hip/hip_runtime.h>
#include <hip/hip_bf16.h>

#define IMGD 128
#define E_   256
#define NH_  8
#define HD_  32
#define WSZ  8
#define L_   4
#define HID_ 1024
#define B_   2
#define NTOK (B_*IMGD*IMGD)   // 32768 tokens
#define TWIN 64               // tokens per window

typedef __attribute__((ext_vector_type(16))) __bf16 bf16x16;
typedef __attribute__((ext_vector_type(8)))  float  f32x8;

__device__ __forceinline__ __bf16 us2bf(unsigned short s) {
    return __builtin_bit_cast(__bf16, s);
}

// packed fp32x2 -> bf16x2 (RNE) via hardware v_cvt_pk_bf16_f32
__device__ __forceinline__ unsigned pk_bf16(float a, float b) {
    unsigned r;
    asm("v_cvt_pk_bf16_f32 %0, %1, %2" : "=v"(r) : "v"(a), "v"(b));
    return r;
}

__device__ __forceinline__ f32x8 wmma_bf16(bf16x16 a, bf16x16 b, f32x8 c) {
    return __builtin_amdgcn_wmma_f32_16x16x32_bf16(false, a, false, b,
                                                   (short)0, c, false, false);
}

// ---------------------------------------------------------------- converts
// 2 elements per thread, packed b32 store
__global__ void convert_bf16_k(const float* __restrict__ in,
                               unsigned* __restrict__ out, int npair) {
    int i = blockIdx.x * 256 + threadIdx.x;
    if (i < npair) out[i] = pk_bf16(in[2 * i], in[2 * i + 1]);
}

// Pack a weight matrix [K,N] (fp32) into per-fragment bf16 order:
// fragment (n-tile tn, k-step kb): 32 lanes x 16 elems contiguous;
// lane l = (kr>>4)*16 + (n&15), elem e = kr&15, value = B[k][n].
// trans=1 reads src as [N,K] (for out_w where B[k][n] = out_w[n][k]).
__global__ void pack_b_k(const float* __restrict__ src,
                         unsigned short* __restrict__ dst,
                         int K, int N, int trans) {
    int i = blockIdx.x * 256 + threadIdx.x;
    if (i >= K * N) return;
    int k = i / N, n = i % N;
    float v = trans ? src[(size_t)n * K + k] : src[(size_t)k * N + n];
    int tile = n >> 4, nl = n & 15;
    int kb = k >> 5, kr = k & 31;
    int lane = ((kr >> 4) << 4) + nl;
    int e = kr & 15;
    dst[(((size_t)tile * (K >> 5) + kb) << 9) + (lane << 4) + e] =
        (unsigned short)pk_bf16(v, v);
}

// ---------------------------------------------------------------- embed
__global__ void embed_k(const float* __restrict__ x, const float* __restrict__ in_w,
                        const float* __restrict__ in_b, const float* __restrict__ pos,
                        float* __restrict__ seq) {
    int token = blockIdx.x;
    int e     = threadIdx.x;
    int b  = token >> 14;
    int hw = token & (IMGD * IMGD - 1);
    const float* xb = x + (size_t)b * 3 * IMGD * IMGD + hw;
    float c0 = xb[0], c1 = xb[IMGD * IMGD], c2 = xb[2 * IMGD * IMGD];
    float v = c0 * in_w[e * 3] + c1 * in_w[e * 3 + 1] + c2 * in_w[e * 3 + 2]
            + in_b[e] + pos[(size_t)e * IMGD * IMGD + hw];
    seq[(size_t)token * E_ + e] = v;
}

// ---------------------------------------------------------------- layernorm
__global__ void __launch_bounds__(256)
layernorm_bf16(const float* __restrict__ x, const float* __restrict__ g,
               const float* __restrict__ b, unsigned short* __restrict__ out) {
    int token = blockIdx.x * 8 + (threadIdx.x >> 5);
    int lane  = threadIdx.x & 31;
    const float* xr = x + (size_t)token * E_ + lane * 8;
    float4 v0 = *(const float4*)xr;
    float4 v1 = *(const float4*)(xr + 4);
    float arr[8] = {v0.x, v0.y, v0.z, v0.w, v1.x, v1.y, v1.z, v1.w};
    float s = 0.f, q = 0.f;
#pragma unroll
    for (int e = 0; e < 8; ++e) { s += arr[e]; q += arr[e] * arr[e]; }
#pragma unroll
    for (int o = 16; o; o >>= 1) { s += __shfl_xor(s, o, 32); q += __shfl_xor(q, o, 32); }
    float mean = s * (1.f / E_);
    float var  = q * (1.f / E_) - mean * mean;
    float rs   = rsqrtf(var + 1e-5f);
    unsigned* orow32 = (unsigned*)(out + (size_t)token * E_ + lane * 8);
    const float* gp = g + lane * 8;
    const float* bp = b + lane * 8;
#pragma unroll
    for (int e = 0; e < 8; e += 2) {
        float a0 = (arr[e]     - mean) * rs * gp[e]     + bp[e];
        float a1 = (arr[e + 1] - mean) * rs * gp[e + 1] + bp[e + 1];
        orow32[e >> 1] = pk_bf16(a0, a1);
    }
}

// ---------------------------------------------------------------- WMMA GEMM
// C = A(bf16 [M,K]) @ Bpacked(bf16, fragment order) + bias
// MODE 0: bf16 store            MODE 1: fp32 residual +=
// MODE 2: bf16 store of GELU    MODE 3: fp32 BCHW transpose-store
template <int MODE>
__global__ void __launch_bounds__(256)
gemm_bf16_wmma(const unsigned short* __restrict__ A,
               const unsigned short* __restrict__ Bp,
               const float* __restrict__ bias,
               float* __restrict__ outF,
               unsigned short* __restrict__ outB,
               int M, int N, int K) {
    int lane = threadIdx.x & 31;
    int wid  = threadIdx.x >> 5;          // 0..7
    int lh = lane >> 4, ll = lane & 15;
    int m0 = blockIdx.y * 64 + (wid >> 1) * 16;
    int tn = blockIdx.x * 4 + (wid & 1) * 2;   // 16-wide n-tile index
    int n0 = tn * 16;
    int KB = K >> 5;

    const unsigned short* Arow = A + (size_t)(m0 + ll) * K + lh * 16;
    const unsigned short* B0 = Bp + (((size_t)tn       * KB) << 9) + lane * 16;
    const unsigned short* B1 = Bp + (((size_t)(tn + 1) * KB) << 9) + lane * 16;

    f32x8 acc0 = {}; f32x8 acc1 = {};
#pragma unroll 4
    for (int kb = 0; kb < KB; ++kb) {
        // A frag: lane=row, elems 0..15 -> K=kb*32+16*lh+e
        union { uint4 u[2]; bf16x16 v; } ua, ub0, ub1;
        ua.u[0]  = *(const uint4*)(Arow + kb * 32);
        ua.u[1]  = *(const uint4*)(Arow + kb * 32 + 8);
        // B frags: pre-packed, 32B contiguous per lane (1KB contiguous per wave)
        ub0.u[0] = *(const uint4*)(B0 + ((size_t)kb << 9));
        ub0.u[1] = *(const uint4*)(B0 + ((size_t)kb << 9) + 8);
        ub1.u[0] = *(const uint4*)(B1 + ((size_t)kb << 9));
        ub1.u[1] = *(const uint4*)(B1 + ((size_t)kb << 9) + 8);
        acc0 = wmma_bf16(ua.v, ub0.v, acc0);
        acc1 = wmma_bf16(ua.v, ub1.v, acc1);
    }
    int n_lo = n0 + ll, n_hi = n0 + 16 + ll;
    float b0 = bias[n_lo], b1 = bias[n_hi];
#pragma unroll
    for (int r = 0; r < 8; ++r) {
        int m = m0 + r + 8 * lh;          // C layout: VGPR r -> row r + 8*(lane>>4)
        float v0 = acc0[r] + b0;
        float v1 = acc1[r] + b1;
        if (MODE == 0) {
            unsigned pk = pk_bf16(v0, v1);
            outB[(size_t)m * N + n_lo] = (unsigned short)pk;
            outB[(size_t)m * N + n_hi] = (unsigned short)(pk >> 16);
        } else if (MODE == 1) {
            outF[(size_t)m * N + n_lo] += v0;
            outF[(size_t)m * N + n_hi] += v1;
        } else if (MODE == 2) {
            v0 = 0.5f * v0 * (1.f + erff(v0 * 0.70710678118654752f));
            v1 = 0.5f * v1 * (1.f + erff(v1 * 0.70710678118654752f));
            unsigned pk = pk_bf16(v0, v1);
            outB[(size_t)m * N + n_lo] = (unsigned short)pk;
            outB[(size_t)m * N + n_hi] = (unsigned short)(pk >> 16);
        } else { // MODE 3: [token][e] -> [b][e][hw] fp32
            int b  = m >> 14;
            int hw = m & (IMGD * IMGD - 1);
            outF[((size_t)b * E_ + n_lo) * (IMGD * IMGD) + hw] = v0;
            outF[((size_t)b * E_ + n_hi) * (IMGD * IMGD) + hw] = v1;
        }
    }
}

// ---------------------------------------------------------------- window attention
// one block per window (512 blocks), one wave per head.
// V rows staged to LDS via async global->LDS DMA (ASYNCcnt path).
__global__ void __launch_bounds__(256)
win_attn_k(const unsigned short* __restrict__ qkv,   // [NTOK][768] = [q|k|v][head][32]
           unsigned short* __restrict__ aout,        // [NTOK][256]
           int shifted) {
    __shared__ int tok[TWIN];
    __shared__ int rcode[TWIN];
    __shared__ unsigned short plds[NH_][16][TWIN];   // 16 KB P staging
    __shared__ unsigned short vbuf[TWIN][E_];        // 32 KB V rows (all heads)

    int blk  = blockIdx.x;
    int bimg = blk >> 8;
    int wy   = (blk >> 4) & 15;
    int wx   = blk & 15;
    int sh   = shifted ? (WSZ / 2) : 0;
    if (threadIdx.x < TWIN) {
        int i = threadIdx.x >> 3, j = threadIdx.x & 7;
        int hs  = wy * WSZ + i;           // coord in (implicitly) rolled image
        int wsc = wx * WSZ + j;
        int h = (hs + sh) & (IMGD - 1);   // map back to unrolled seq
        int w = (wsc + sh) & (IMGD - 1);
        tok[threadIdx.x] = bimg * (IMGD * IMGD) + h * IMGD + w;
        int rh = (hs  >= IMGD - WSZ) + (hs  >= IMGD - WSZ / 2);
        int rw = (wsc >= IMGD - WSZ) + (wsc >= IMGD - WSZ / 2);
        rcode[threadIdx.x] = rh * 3 + rw;
    }
    __syncthreads();

    // ---- async DMA: stage the window's 64 V rows (64 x 512B) into LDS ----
    {
        int t = threadIdx.x;
#pragma unroll
        for (int it = 0; it < 8; ++it) {
            int seg = t + it * 256;       // 0..2047 16B-segments
            int row = seg >> 5;           // token slot
            int s16 = seg & 31;           // segment within 512B row
            const unsigned short* g = qkv + (size_t)tok[row] * 768 + 512 + s16 * 8;
            unsigned lds = (unsigned)(size_t)&vbuf[row][s16 * 8];
            unsigned long long ga = (unsigned long long)(size_t)g;
            asm volatile("global_load_async_to_lds_b128 %0, %1, off"
                         :: "v"(lds), "v"(ga) : "memory");
        }
        asm volatile("s_wait_asynccnt 0x0" ::: "memory");
    }
    __syncthreads();

    int head = threadIdx.x >> 5;
    int lane = threadIdx.x & 31;
    int lh = lane >> 4, ll = lane & 15;

    // K fragments (B operand of Q@K^T): 4 s-tiles, contraction dim = d(32)
    bf16x16 kf[4];
#pragma unroll
    for (int st = 0; st < 4; ++st) {
        const unsigned short* kp = qkv + (size_t)tok[st * 16 + ll] * 768
                                 + 256 + head * HD_ + lh * 16;
        union { uint4 u[2]; bf16x16 v; } u;
        u.u[0] = *(const uint4*)kp;
        u.u[1] = *(const uint4*)(kp + 8);
        kf[st] = u.v;
    }
    // V fragments from LDS: contraction dim = s(64) in 2 k-steps, 2 n-tiles
    bf16x16 vf[2][2];
#pragma unroll
    for (int ks = 0; ks < 2; ++ks)
#pragma unroll
        for (int e = 0; e < 16; ++e) {
            int s = ks * 32 + lh * 16 + e;
            vf[ks][0][e] = us2bf(vbuf[s][head * HD_ + ll]);
            vf[ks][1][e] = us2bf(vbuf[s][head * HD_ + 16 + ll]);
        }

    const float scale = 0.17677669529663689f;  // 1/sqrt(32)
#pragma unroll 1
    for (int strip = 0; strip < 4; ++strip) {
        const unsigned short* qp = qkv + (size_t)tok[strip * 16 + ll] * 768
                                 + head * HD_ + lh * 16;
        union { uint4 u[2]; bf16x16 v; } qa;
        qa.u[0] = *(const uint4*)qp;
        qa.u[1] = *(const uint4*)(qp + 8);

        f32x8 sacc[4];
#pragma unroll
        for (int st = 0; st < 4; ++st) {
            sacc[st] = {};
            sacc[st] = wmma_bf16(qa.v, kf[st], sacc[st]);
        }
        float vals[4][8];
#pragma unroll
        for (int st = 0; st < 4; ++st) {
            int cs = rcode[st * 16 + ll];
#pragma unroll
            for (int r = 0; r < 8; ++r) {
                float v = sacc[st][r] * scale;
                if (shifted) {
                    int ct = rcode[strip * 16 + r + 8 * lh];
                    v += (ct != cs) ? -100.f : 0.f;
                }
                vals[st][r] = v;
            }
        }
#pragma unroll
        for (int r = 0; r < 8; ++r) {
            float mx = fmaxf(fmaxf(vals[0][r], vals[1][r]),
                             fmaxf(vals[2][r], vals[3][r]));
#pragma unroll
            for (int o = 8; o; o >>= 1) mx = fmaxf(mx, __shfl_xor(mx, o, 32));
            float sm = 0.f;
#pragma unroll
            for (int st = 0; st < 4; ++st) {
                float e = __expf(vals[st][r] - mx);
                vals[st][r] = e; sm += e;
            }
#pragma unroll
            for (int o = 8; o; o >>= 1) sm += __shfl_xor(sm, o, 32);
            float inv = 1.f / sm;
            unsigned p01 = pk_bf16(vals[0][r] * inv, vals[1][r] * inv);
            unsigned p23 = pk_bf16(vals[2][r] * inv, vals[3][r] * inv);
            unsigned short* pr = &plds[head][r + 8 * lh][ll];
            pr[0]  = (unsigned short)p01;
            pr[16] = (unsigned short)(p01 >> 16);
            pr[32] = (unsigned short)p23;
            pr[48] = (unsigned short)(p23 >> 16);
        }
        // reload P as A-frags from LDS (C layout -> A layout transpose in-wave)
        bf16x16 pf[2];
#pragma unroll
        for (int ks = 0; ks < 2; ++ks)
#pragma unroll
            for (int e = 0; e < 16; ++e) {
                int sidx = ks * 32 + lh * 16 + e;
                pf[ks][e] = us2bf(plds[head][ll][sidx]);
            }
        f32x8 oa0 = {}, oa1 = {};
        oa0 = wmma_bf16(pf[0], vf[0][0], oa0);
        oa0 = wmma_bf16(pf[1], vf[1][0], oa0);
        oa1 = wmma_bf16(pf[0], vf[0][1], oa1);
        oa1 = wmma_bf16(pf[1], vf[1][1], oa1);
#pragma unroll
        for (int r = 0; r < 8; ++r) {
            int t = tok[strip * 16 + r + 8 * lh];
            unsigned short* op = aout + (size_t)t * E_ + head * HD_;
            unsigned pk = pk_bf16(oa0[r], oa1[r]);
            op[ll]      = (unsigned short)pk;
            op[16 + ll] = (unsigned short)(pk >> 16);
        }
    }
}

// ---------------------------------------------------------------- launch
extern "C" void kernel_launch(void* const* d_in, const int* in_sizes, int n_in,
                              void* d_out, int out_size, void* d_ws, size_t ws_size,
                              hipStream_t stream) {
    const float* x      = (const float*)d_in[0];
    const float* in_w   = (const float*)d_in[1];
    const float* in_b   = (const float*)d_in[2];
    const float* pos    = (const float*)d_in[3];
    const float* ln1_g  = (const float*)d_in[4];
    const float* ln1_b  = (const float*)d_in[5];
    const float* qkv_w  = (const float*)d_in[6];
    const float* qkv_b  = (const float*)d_in[7];
    const float* proj_w = (const float*)d_in[8];
    const float* proj_b = (const float*)d_in[9];
    const float* ln2_g  = (const float*)d_in[10];
    const float* ln2_b  = (const float*)d_in[11];
    const float* mlp_w1 = (const float*)d_in[12];
    const float* mlp_b1 = (const float*)d_in[13];
    const float* mlp_w2 = (const float*)d_in[14];
    const float* mlp_b2 = (const float*)d_in[15];
    const float* out_w  = (const float*)d_in[16];
    const float* out_b  = (const float*)d_in[17];

    char* p = (char*)d_ws;
    auto alloc = [&](size_t bytes) -> char* {
        char* r = p;
        p += (bytes + 255) & ~(size_t)255;
        return r;
    };
    float*          seq  = (float*)alloc((size_t)NTOK * E_ * 4);            // 33.5 MB
    unsigned short* hbuf = (unsigned short*)alloc((size_t)NTOK * E_ * 2);   // 16.8 MB
    unsigned short* big  = (unsigned short*)alloc((size_t)NTOK * HID_ * 2); // qkv/hidden
    unsigned short* wq = (unsigned short*)alloc((size_t)L_ * E_ * 3 * E_ * 2);
    unsigned short* wp = (unsigned short*)alloc((size_t)L_ * E_ * E_ * 2);
    unsigned short* w1 = (unsigned short*)alloc((size_t)L_ * E_ * HID_ * 2);
    unsigned short* w2 = (unsigned short*)alloc((size_t)L_ * HID_ * E_ * 2);
    unsigned short* wo = (unsigned short*)alloc((size_t)E_ * E_ * 2);

    // one-time weight packing into WMMA fragment order
    for (int l = 0; l < L_; ++l) {
        int n;
        n = E_ * 3 * E_;
        pack_b_k<<<(n + 255) / 256, 256, 0, stream>>>(
            qkv_w + (size_t)l * n, wq + (size_t)l * n, E_, 3 * E_, 0);
        n = E_ * E_;
        pack_b_k<<<(n + 255) / 256, 256, 0, stream>>>(
            proj_w + (size_t)l * n, wp + (size_t)l * n, E_, E_, 0);
        n = E_ * HID_;
        pack_b_k<<<(n + 255) / 256, 256, 0, stream>>>(
            mlp_w1 + (size_t)l * n, w1 + (size_t)l * n, E_, HID_, 0);
        n = HID_ * E_;
        pack_b_k<<<(n + 255) / 256, 256, 0, stream>>>(
            mlp_w2 + (size_t)l * n, w2 + (size_t)l * n, HID_, E_, 0);
    }
    pack_b_k<<<(E_ * E_ + 255) / 256, 256, 0, stream>>>(out_w, wo, E_, E_, 1);

    embed_k<<<NTOK, 256, 0, stream>>>(x, in_w, in_b, pos, seq);

    for (int l = 0; l < L_; ++l) {
        layernorm_bf16<<<NTOK / 8, 256, 0, stream>>>(seq, ln1_g + l * E_, ln1_b + l * E_, hbuf);
        gemm_bf16_wmma<0><<<dim3((3 * E_) / 64, NTOK / 64), 256, 0, stream>>>(
            hbuf, wq + (size_t)l * E_ * 3 * E_, qkv_b + l * 3 * E_,
            nullptr, big, NTOK, 3 * E_, E_);
        win_attn_k<<<B_ * 16 * 16, 256, 0, stream>>>(big, hbuf, l & 1);
        gemm_bf16_wmma<1><<<dim3(E_ / 64, NTOK / 64), 256, 0, stream>>>(
            hbuf, wp + (size_t)l * E_ * E_, proj_b + l * E_,
            seq, nullptr, NTOK, E_, E_);
        layernorm_bf16<<<NTOK / 8, 256, 0, stream>>>(seq, ln2_g + l * E_, ln2_b + l * E_, hbuf);
        gemm_bf16_wmma<2><<<dim3(HID_ / 64, NTOK / 64), 256, 0, stream>>>(
            hbuf, w1 + (size_t)l * E_ * HID_, mlp_b1 + l * HID_,
            nullptr, big, NTOK, HID_, E_);
        gemm_bf16_wmma<1><<<dim3(E_ / 64, NTOK / 64), 256, 0, stream>>>(
            big, w2 + (size_t)l * HID_ * E_, mlp_b2 + l * E_,
            seq, nullptr, NTOK, E_, HID_);
    }

    int npair = (NTOK * E_) / 2;
    convert_bf16_k<<<(npair + 255) / 256, 256, 0, stream>>>(seq, (unsigned*)hbuf, npair);
    gemm_bf16_wmma<3><<<dim3(E_ / 64, NTOK / 64), 256, 0, stream>>>(
        hbuf, wo, out_b, (float*)d_out, nullptr, NTOK, E_, E_);
}
